// SelfAttention_43233140801717
// MI455X (gfx1250) — matrix operand order repaired and checked by reading, not compiled, not run on hardware
//
#include <hip/hip_runtime.h>
#include <hip/hip_bf16.h>

typedef __attribute__((ext_vector_type(8)))  float  v8f;
typedef __attribute__((ext_vector_type(16))) __bf16 v16bf;
typedef __attribute__((ext_vector_type(8)))  __bf16 v8bf;

#define TOK   4096
#define CIN   4096
#define DD    32
#define NROWS 16384   // 4 batches * 4096 tokens

static __device__ __forceinline__ v16bf ld16bf(const __bf16* p) {
  return *(const v16bf*)p;   // 32B vector load (2 x b128)
}

// DETR sine positional embedding value for (token, channel)
static __device__ __forceinline__ float posval(int token, int n) {
  int h = token >> 7;        // PW = 128
  int w = token & 127;
  float v = (n < 16) ? (float)(h + 1) : (float)(w + 1);
  int i = n & 15;
  float e = (float)(i >> 1) * 0.125f;          // floor(i/2)/8
  float denom = __powf(10000.0f, e);
  float arg = v / denom;
  return (i & 1) ? __cosf(arg) : __sinf(arg);
}

// ---------------------------------------------------------------------------
// Kernel 0: pack W_emb (4096 x 32, f32, row-major) into bf16 B-fragment layout
// ---------------------------------------------------------------------------
__global__ void pack_wemb_kernel(const float* __restrict__ W,
                                 __bf16* __restrict__ Wb) {
  int idx  = blockIdx.x * blockDim.x + threadIdx.x;   // 131072 total
  int i    = idx & 15;
  int lane = (idx >> 4) & 31;
  int kb   = (idx >> 9) & 127;
  int nb   = (idx >> 16) & 1;
  int n = nb * 16 + (lane & 15);
  int k = kb * 32 + ((lane >> 4) ? 8 : 0) + (i < 8 ? i : i + 8);
  Wb[idx] = (__bf16)W[k * DD + n];
}

// ---------------------------------------------------------------------------
// Kernel 1: X = LN(corr @ W_emb + b_emb) + pos   (the 256 MB streaming GEMM)
// One wave per 16-row tile; 2 bf16 WMMAs per 32-wide K step, 128 steps.
// ---------------------------------------------------------------------------
__global__ void embed_kernel(const float* __restrict__ corr,
                             const __bf16* __restrict__ Wb,
                             const float* __restrict__ b_emb,
                             const float* __restrict__ gamma,
                             const float* __restrict__ beta,
                             float* __restrict__ X) {
  const int lane = threadIdx.x & 31;
  const int half = lane >> 4;
  const int c    = lane & 15;
  const int tile = (blockIdx.x * blockDim.x + threadIdx.x) >> 5;  // 1024 tiles
  const int r0   = tile * 16;                                     // global row
  const float* arow = corr + (size_t)(r0 + c) * CIN;
  const int aoff = half * 8;   // lanes 0-15: K 0-7 & 16-23; lanes 16-31: 8-15 & 24-31

  v8f acc0 = {};
  v8f acc1 = {};

  for (int ks = 0; ks < 128; ++ks) {
    const int kb = ks * 32;
    float4 a0 = *(const float4*)(arow + kb + aoff);
    float4 a1 = *(const float4*)(arow + kb + aoff + 4);
    float4 a2 = *(const float4*)(arow + kb + aoff + 16);
    float4 a3 = *(const float4*)(arow + kb + aoff + 20);
    __builtin_prefetch(arow + kb + 128, 0, 1);   // global_prefetch_b8 path
    v16bf af;
    af[0]=(__bf16)a0.x; af[1]=(__bf16)a0.y; af[2]=(__bf16)a0.z; af[3]=(__bf16)a0.w;
    af[4]=(__bf16)a1.x; af[5]=(__bf16)a1.y; af[6]=(__bf16)a1.z; af[7]=(__bf16)a1.w;
    af[8]=(__bf16)a2.x; af[9]=(__bf16)a2.y; af[10]=(__bf16)a2.z; af[11]=(__bf16)a2.w;
    af[12]=(__bf16)a3.x; af[13]=(__bf16)a3.y; af[14]=(__bf16)a3.z; af[15]=(__bf16)a3.w;
    v16bf b0 = ld16bf(Wb + ((size_t)(ks)       * 32 + lane) * 16);
    v16bf b1 = ld16bf(Wb + ((size_t)(128 + ks) * 32 + lane) * 16);
    acc0 = __builtin_amdgcn_wmma_f32_16x16x32_bf16(false, af, false, b0,
                                                   (short)0, acc0, false, false);
    acc1 = __builtin_amdgcn_wmma_f32_16x16x32_bf16(false, af, false, b1,
                                                   (short)0, acc1, false, false);
  }

  const float bia0 = b_emb[c],      bia1 = b_emb[16 + c];
  const float g0   = gamma[c],      g1   = gamma[16 + c];
  const float be0  = beta[c],       be1  = beta[16 + c];

#pragma unroll
  for (int j = 0; j < 8; ++j) {
    float x0 = acc0[j] + bia0;
    float x1 = acc1[j] + bia1;
    float s1 = x0 + x1;
    float s2 = x0 * x0 + x1 * x1;
    s1 += __shfl_xor(s1, 1); s1 += __shfl_xor(s1, 2);
    s1 += __shfl_xor(s1, 4); s1 += __shfl_xor(s1, 8);
    s2 += __shfl_xor(s2, 1); s2 += __shfl_xor(s2, 2);
    s2 += __shfl_xor(s2, 4); s2 += __shfl_xor(s2, 8);
    float mu  = s1 * (1.0f / 32.0f);
    float var = s2 * (1.0f / 32.0f) - mu * mu;
    float rs  = rsqrtf(var + 1e-5f);
    int rloc  = half ? (j + 8) : j;
    int row   = r0 + rloc;
    int token = row & (TOK - 1);
    float y0 = (x0 - mu) * rs * g0 + be0 + posval(token, c);
    float y1 = (x1 - mu) * rs * g1 + be1 + posval(token, 16 + c);
    X[(size_t)row * DD + c]      = y0;
    X[(size_t)row * DD + 16 + c] = y1;
  }
}

// ---------------------------------------------------------------------------
// Kernel 2: QKV projections (tiny, fp32). Q (pre-scaled by 1/scale) and K
// row-major bf16; V transposed (batch, d, token).
// ---------------------------------------------------------------------------
__global__ void qkv_kernel(const float* __restrict__ X,
                           const float* __restrict__ Wq, const float* __restrict__ bq,
                           const float* __restrict__ Wk, const float* __restrict__ bk,
                           const float* __restrict__ Wv, const float* __restrict__ bv,
                           const float* __restrict__ scale_p,
                           __bf16* __restrict__ Q, __bf16* __restrict__ K,
                           __bf16* __restrict__ Vt) {
  int idx = blockIdx.x * blockDim.x + threadIdx.x;   // 524288
  int row = idx >> 5;
  int n   = idx & 31;
  float xv = X[(size_t)row * DD + n];                // lane n holds X[row][n]
  float aq = bq[n], ak = bk[n], av = bv[n];
#pragma unroll
  for (int d = 0; d < 32; ++d) {
    float xd = __shfl(xv, d, 32);
    aq = fmaf(xd, Wq[d * DD + n], aq);
    ak = fmaf(xd, Wk[d * DD + n], ak);
    av = fmaf(xd, Wv[d * DD + n], av);
  }
  float inv_scale = 1.0f / scale_p[0];               // fold softmax scale into Q
  Q[(size_t)row * DD + n] = (__bf16)(aq * inv_scale);
  K[(size_t)row * DD + n] = (__bf16)ak;
  int b = row >> 12, t = row & (TOK - 1);
  Vt[((size_t)b * DD + n) * TOK + t] = (__bf16)av;
}

// ---------------------------------------------------------------------------
// Kernel 3: flash attention, TRANSPOSED formulation.
//   S^T = K_tile @ Q^T  -> each lane holds scores of ONE query (its column),
//   so softmax max/sum are in-register folds + a single xor-16 shuffle.
//   O^T = V^T @ P^T     -> A-fragments load contiguously from d-major Vt.
// One wave per 16 queries; 128 key-blocks of 32; no LDS.
// ---------------------------------------------------------------------------
__global__ void attn_kernel(const __bf16* __restrict__ Q,
                            const __bf16* __restrict__ K,
                            const __bf16* __restrict__ Vt,
                            float* __restrict__ out) {
  const int lane = threadIdx.x & 31;
  const int half = lane >> 4;
  const int c    = lane & 15;
  const int tile = blockIdx.x * (blockDim.x >> 5) + (threadIdx.x >> 5); // 1024
  const int R = tile * 16;                   // global q row base
  const int b = R >> 12;
  const int aoff = half * 16;

  // B-fragment of Q^T: lane -> query column c, contiguous d (half*16 .. +15)
  v16bf qb = ld16bf(Q + (size_t)(R + c) * DD + half * 16);

  const __bf16* Kbase  = K  + (size_t)b * TOK * DD;
  const __bf16* Vtbase = Vt + (size_t)b * DD * TOK;

  // A-fragment row pointers (lane's row): keys for K, d-rows for Vt
  const __bf16* krow0 = Kbase + (size_t)c * DD;          // keys c      (+kb*DD)
  const __bf16* krow1 = Kbase + (size_t)(16 + c) * DD;   // keys 16+c
  const __bf16* vrow0 = Vtbase + (size_t)c * TOK;        // d = c
  const __bf16* vrow1 = Vtbase + (size_t)(16 + c) * TOK; // d = 16+c

  v8f o0 = {}, o1 = {};          // O^T accumulators: d-tiles 0 and 1
  float m = -1e30f, lsum = 0.0f; // flash state for THIS lane's query

  for (int kb = 0; kb < TOK; kb += 32) {
    // A-fragments of K (16 keys x 32 d) for key sub-tiles 0..15 / 16..31
    v16bf ka0, ka1;
    {
      v8bf k0lo = *(const v8bf*)(krow0 + (size_t)kb * DD + aoff);
      v8bf k0hi = *(const v8bf*)(krow0 + (size_t)kb * DD + aoff + 8);
      v8bf k1lo = *(const v8bf*)(krow1 + (size_t)kb * DD + aoff);
      v8bf k1hi = *(const v8bf*)(krow1 + (size_t)kb * DD + aoff + 8);
#pragma unroll
      for (int i = 0; i < 8; ++i) {
        ka0[i] = k0lo[i]; ka0[8 + i] = k0hi[i];
        ka1[i] = k1lo[i]; ka1[8 + i] = k1hi[i];
      }
    }
    v8f s0 = {}, s1 = {};
    s0 = __builtin_amdgcn_wmma_f32_16x16x32_bf16(false, ka0, false, qb,
                                                 (short)0, s0, false, false);
    s1 = __builtin_amdgcn_wmma_f32_16x16x32_bf16(false, ka1, false, qb,
                                                 (short)0, s1, false, false);

    // ---- online softmax (scores for this lane's query live in s0/s1) ----
    float mx = fmaxf(s0[0], s1[0]);
#pragma unroll
    for (int j = 1; j < 8; ++j) mx = fmaxf(mx, fmaxf(s0[j], s1[j]));
    mx = fmaxf(mx, __shfl_xor(mx, 16));       // merge the two key halves
    float mnew = fmaxf(m, mx);
    float cf = __expf(m - mnew);
    m = mnew;

    float p0[8], p1[8];
    float ps = 0.0f;
#pragma unroll
    for (int j = 0; j < 8; ++j) {
      p0[j] = __expf(s0[j] - mnew);
      p1[j] = __expf(s1[j] - mnew);
      ps += p0[j] + p1[j];
    }
    ps += __shfl_xor(ps, 16);
    lsum = lsum * cf + ps;
#pragma unroll
    for (int j = 0; j < 8; ++j) { o0[j] *= cf; o1[j] *= cf; }

    // ---- assemble B-fragment of P^T (32 keys x 16 q) via xor-16 swap ----
    // lane<16 holds keys {j, 16+j}; lane>=16 holds keys {8+j, 24+j}
    v16bf pb;
#pragma unroll
    for (int j = 0; j < 8; ++j) {
      float t0 = __shfl_xor(p0[j], 16);   // other half's tile0 value
      float t1 = __shfl_xor(p1[j], 16);   // other half's tile1 value
      float lo = half ? t1 : p0[j];       // keys base+16*half+0..7 for this lane
      float hi = half ? p1[j] : t0;       // keys base+16*half+8..15 for this lane
      pb[j]     = (__bf16)lo;
      pb[8 + j] = (__bf16)hi;
    }

    // A-fragments of V^T (16 d x 32 keys), d-tiles 0 and 1
    v16bf va0, va1;
    {
      v8bf v0lo = *(const v8bf*)(vrow0 + kb + aoff);
      v8bf v0hi = *(const v8bf*)(vrow0 + kb + aoff + 8);
      v8bf v1lo = *(const v8bf*)(vrow1 + kb + aoff);
      v8bf v1hi = *(const v8bf*)(vrow1 + kb + aoff + 8);
#pragma unroll
      for (int i = 0; i < 8; ++i) {
        va0[i] = v0lo[i]; va0[8 + i] = v0hi[i];
        va1[i] = v1lo[i]; va1[8 + i] = v1hi[i];
      }
    }
    o0 = __builtin_amdgcn_wmma_f32_16x16x32_bf16(false, va0, false, pb,
                                                 (short)0, o0, false, false);
    o1 = __builtin_amdgcn_wmma_f32_16x16x32_bf16(false, va1, false, pb,
                                                 (short)0, o1, false, false);
  }

  // O^T D-layout: element (j,lane): d = tile*16 + (half? j+8 : j), q = c
  float inv = 1.0f / lsum;
  float* orow = out + (size_t)(R + c) * DD;
#pragma unroll
  for (int j = 0; j < 8; ++j) {
    int d0 = half ? (j + 8) : j;
    orow[d0]      = o0[j] * inv;
    orow[16 + d0] = o1[j] * inv;
  }
}

// ---------------------------------------------------------------------------
extern "C" void kernel_launch(void* const* d_in, const int* in_sizes, int n_in,
                              void* d_out, int out_size, void* d_ws, size_t ws_size,
                              hipStream_t stream) {
  const float* corr  = (const float*)d_in[0];
  const float* W_emb = (const float*)d_in[1];
  const float* b_emb = (const float*)d_in[2];
  const float* ln_g  = (const float*)d_in[3];
  const float* ln_b  = (const float*)d_in[4];
  const float* Wq = (const float*)d_in[5];  const float* bq = (const float*)d_in[6];
  const float* Wk = (const float*)d_in[7];  const float* bk = (const float*)d_in[8];
  const float* Wv = (const float*)d_in[9];  const float* bv = (const float*)d_in[10];
  const float* scale = (const float*)d_in[11];

  char* ws = (char*)d_ws;
  __bf16* Wb = (__bf16*)ws;                                  // 256 KB
  float*  X  = (float*)(ws + (256u << 10));                  // 2 MB
  __bf16* Q  = (__bf16*)(ws + (256u << 10) + (2u << 20));    // 1 MB
  __bf16* K  = (__bf16*)(ws + (256u << 10) + (3u << 20));    // 1 MB
  __bf16* Vt = (__bf16*)(ws + (256u << 10) + (4u << 20));    // 1 MB
  float* out = (float*)d_out;

  pack_wemb_kernel<<<512, 256, 0, stream>>>(W_emb, Wb);
  embed_kernel<<<128, 256, 0, stream>>>(corr, Wb, b_emb, ln_g, ln_b, X);
  qkv_kernel<<<2048, 256, 0, stream>>>(X, Wq, bq, Wk, bk, Wv, bv, scale, Q, K, Vt);
  attn_kernel<<<128, 256, 0, stream>>>(Q, K, Vt, out);
}
